// GraphAttentionLayer_52484500357558
// MI455X (gfx1250) — matrix operand order repaired and checked by reading, compiled, not run on hardware
//
#include <hip/hip_runtime.h>

typedef __attribute__((ext_vector_type(2)))  float    v2f;
typedef __attribute__((ext_vector_type(8)))  float    v8f;
typedef __attribute__((ext_vector_type(8)))  _Float16 v8h;
typedef __attribute__((ext_vector_type(16))) _Float16 v16h;

// Problem dims (fixed by the reference)
constexpr int B_   = 8;
constexpr int N_   = 1024;
constexpr int FIN  = 256;
constexpr int FOUT = 256;
constexpr int NOA  = 4;
constexpr int H_   = 5;     // noA + identity head
constexpr int FSQ  = 32;
constexpr int ROWS = B_ * N_;          // 8192 flattened (b,n) rows
constexpr int KD   = H_ * FSQ;         // 160 concat width

#define NEG_BIG (-9e15f)
#define SLOPE   0.01f
#define LN_EPS  1e-5f

__device__ __forceinline__ float lrelu(float x) { return x > 0.f ? x : SLOPE * x; }

// ---------------------------------------------------------------------------
// Kernel P: transpose weights so WMMA B-fragments become contiguous b64 loads.
//   WT[k][n][kin]   = W[k][kin][n]        (H*FSQ*FIN)
//   sq_wT[col][kin] = sq_w[kin][col]      (FOUT*KD)
// ---------------------------------------------------------------------------
__global__ void prep_kernel(const float* __restrict__ W, const float* __restrict__ sq_w,
                            float* __restrict__ WT, float* __restrict__ sq_wT) {
  const int t = blockIdx.x * blockDim.x + threadIdx.x;
  constexpr int nW = H_ * FSQ * FIN;   // 40960
  constexpr int nS = FOUT * KD;        // 40960
  if (t < nW) {
    const int k   = t / (FSQ * FIN);
    const int rem = t % (FSQ * FIN);
    const int n   = rem / FIN;
    const int kin = rem % FIN;
    WT[t] = W[((size_t)k * FIN + kin) * FSQ + n];
  } else if (t < nW + nS) {
    const int o   = t - nW;
    const int col = o / KD;
    const int kin = o % KD;
    sq_wT[o] = sq_w[(size_t)kin * FOUT + col];
  }
}

// ---------------------------------------------------------------------------
// Kernel A: h[k] = V @ W[k]   (f32 WMMA 16x16x4, each wave does a 16x32 strip)
// Also emits h16t[k][b][col][n] — an f16 *transposed* copy of h so the
// attention GEMM's B-fragments are single contiguous 32B loads.
// grid = H_*64 blocks of 256 threads (8 waves)
// ---------------------------------------------------------------------------
__global__ __launch_bounds__(256) void head_gemm_kernel(
    const float* __restrict__ V, const float* __restrict__ WT,
    float* __restrict__ h, _Float16* __restrict__ h16t) {
  const int wave = threadIdx.x >> 5;
  const int lane = threadIdx.x & 31;
  const int half = lane >> 4;
  const int l16  = lane & 15;

  const int k      = blockIdx.x >> 6;     // head
  const int mblock = blockIdx.x & 63;
  const int mtile  = mblock * 8 + wave;
  const int m0     = mtile * 16;

  const float* WkT  = WT + (size_t)k * FSQ * FIN;
  const float* arow = V + (size_t)(m0 + l16) * FIN + 2 * half;
  const float* b0p  = WkT + (size_t)l16 * FIN + 2 * half;        // N = l16
  const float* b1p  = WkT + (size_t)(16 + l16) * FIN + 2 * half; // N = 16+l16

  v8f c0 = {}; v8f c1 = {};
  for (int kb = 0; kb < FIN; kb += 4) {
    const v2f a  = *(const v2f*)(arow + kb);   // A[m][kb+2h], +1
    const v2f b0 = *(const v2f*)(b0p + kb);    // B[kb+2h][n], B[kb+2h+1][n]
    const v2f b1 = *(const v2f*)(b1p + kb);
    c0 = __builtin_amdgcn_wmma_f32_16x16x4_f32(false, a, false, b0, (short)0, c0, false, false);
    c1 = __builtin_amdgcn_wmma_f32_16x16x4_f32(false, a, false, b1, (short)0, c1, false, false);
  }

  float* hk = h + (size_t)k * ROWS * FSQ;
  #pragma unroll
  for (int r = 0; r < 8; ++r) {
    const int row = m0 + r + half * 8;
    hk[(size_t)row * FSQ + l16]      = c0[r];
    hk[(size_t)row * FSQ + 16 + l16] = c1[r];
    const int bb = row >> 10;           // batch
    const int nn = row & (N_ - 1);      // node
    _Float16* ht = h16t + ((size_t)k * B_ + bb) * FSQ * N_;
    ht[(size_t)l16 * N_ + nn]        = (_Float16)c0[r];
    ht[(size_t)(16 + l16) * N_ + nn] = (_Float16)c1[r];
  }
}

// ---------------------------------------------------------------------------
// Kernel B: f1/f2 = h @ a-halves (one thread per (k,b,n) row, K=32 dot)
// ---------------------------------------------------------------------------
__global__ void score_kernel(const float* __restrict__ h, const float* __restrict__ a,
                             float* __restrict__ f1, float* __restrict__ f2) {
  const int row = blockIdx.x * blockDim.x + threadIdx.x;
  if (row >= H_ * ROWS) return;
  const int k = row / ROWS;
  const float* hr = h + (size_t)row * FSQ;
  const float* ak = a + (size_t)k * 2 * FSQ;
  float s1 = 0.f, s2 = 0.f;
  #pragma unroll
  for (int c = 0; c < FSQ; ++c) {
    const float v = hr[c];
    s1 += v * ak[c];
    s2 += v * ak[FSQ + c];
  }
  f1[row] = s1;
  f2[row] = s2;
}

// ---------------------------------------------------------------------------
// Kernel C: fused masked-softmax + agg = att@h + LayerNorm + leaky -> y
// One block = (head k, batch b, 16-row block). 8 waves:
//   stage1: each wave owns 2 rows -> exp tile (f16) in LDS + 1/rowsum
//           (adj stream prefetched ahead; 134MB = the bandwidth roofline)
//   stage2: waves split (ntile x Kquarter); f16 WMMA 16x16x32
//           A from LDS via 2x ds_load_b128, B from h16t via one 32B load
//   stage3: LN over FSQ=32 per row, write concat buffer y
// ---------------------------------------------------------------------------
__global__ __launch_bounds__(256) void attn_kernel(
    const _Float16* __restrict__ h16t, const int* __restrict__ adj,
    const float* __restrict__ f1, const float* __restrict__ f2,
    const float* __restrict__ ln_w, const float* __restrict__ ln_b,
    float* __restrict__ y) {
  __shared__ alignas(32) _Float16 attS[16][N_ + 8]; // 2064B rows: 16B-aligned, 4-bank rotate
  __shared__ float invS[16];
  __shared__ float aggS[16][FSQ];

  const int tid  = threadIdx.x;
  const int wave = tid >> 5;
  const int lane = tid & 31;
  const int half = lane >> 4;
  const int l16  = lane & 15;

  const int k      = blockIdx.x / (B_ * (N_ / 16));   // /512
  const int rem    = blockIdx.x % (B_ * (N_ / 16));
  const int b      = rem >> 6;
  const int iblock = rem & 63;

  for (int t = tid; t < 16 * FSQ; t += blockDim.x) ((float*)aggS)[t] = 0.f;

  const float* f1k = f1 + (size_t)k * ROWS + (size_t)b * N_;
  const float* f2k = f2 + (size_t)k * ROWS + (size_t)b * N_;
  const _Float16* hkt = h16t + ((size_t)k * B_ + b) * FSQ * N_;

  // ---- stage 1: per-row masked softmax numerators (raw exp) ----
  for (int which = 0; which < 2; ++which) {
    const int rl = wave * 2 + which;        // local row 0..15
    const int i  = iblock * 16 + rl;        // row in [0, N)

    float cpre0 = 0.f, cpre1 = 0.f;
    if (i < N_ / 2) {   // closed form: constant per half-row
      cpre0 = lrelu(f1k[2 * i]     + f2k[2 * i]);
      cpre1 = lrelu(f1k[2 * i + 1] + f2k[2 * i + 1]);
    }
    const int* adjrow = adj + (((size_t)(b * N_ + i)) * NOA + (size_t)k) * N_;

    float vals[32];
    float rmax = NEG_BIG;
    #pragma unroll
    for (int it = 0; it < 32; ++it) {
      const int j = it * 32 + lane;
      if (it + 8 < 32) __builtin_prefetch(adjrow + j + 256, 0, 1); // 1KB ahead
      float e;
      if (i < N_ / 2) {
        e = (j >= N_ / 2) ? cpre1 : cpre0;
      } else {          // closed form: independent of i
        e = lrelu(f1k[(2 * j) & (N_ - 1)] + f2k[(2 * j + 1) & (N_ - 1)]);
      }
      const bool m = (k < NOA) ? (adjrow[j] > 0) : (j == i);
      const float v = m ? e : NEG_BIG;
      vals[it] = v;
      rmax = fmaxf(rmax, v);
    }
    #pragma unroll
    for (int o = 16; o > 0; o >>= 1) rmax = fmaxf(rmax, __shfl_xor(rmax, o, 32));

    float sum = 0.f;
    #pragma unroll
    for (int it = 0; it < 32; ++it) {
      const int j = it * 32 + lane;
      const float s = __expf(vals[it] - rmax);   // all-masked row -> all 1 (uniform)
      sum += s;
      attS[rl][j] = (_Float16)s;                 // s in [0,1]: f16-safe
    }
    #pragma unroll
    for (int o = 16; o > 0; o >>= 1) sum += __shfl_xor(sum, o, 32);
    if (lane == 0) invS[rl] = 1.f / sum;
  }
  __syncthreads();

  // ---- stage 2: agg(16x32) = attS(16x1024) @ h(1024x32), f16 WMMA ----
  {
    const int ntile = wave & 1;       // output cols 0-15 / 16-31
    const int kq    = wave >> 1;      // K quarter 0..3 (256 each)
    const int n0    = ntile * 16;

    v8f c = {};
    for (int t = 0; t < 8; ++t) {
      const int jb = kq * 256 + t * 32;
      // A 16x32 f16 fragment = two contiguous 8-half runs per lane:
      //   elements 0..7  <- K = jb + 16*half   .. +7   (VGPR 0..3)
      //   elements 8..15 <- K = jb+16*half+8  .. +7   (VGPR 4..7)
      const v8h alo = *(const v8h*)(&attS[l16][jb + 16 * half]);
      const v8h ahi = *(const v8h*)(&attS[l16][jb + 16 * half + 8]);
      v16h a;
      #pragma unroll
      for (int q = 0; q < 8; ++q) { a[q] = alo[q]; a[8 + q] = ahi[q]; }
      // B 32x16 f16 fragment: K = jb+16*half .. +15 at fixed col -> contiguous
      const v16h bb = *(const v16h*)(hkt + (size_t)(n0 + l16) * N_ + jb + 16 * half);
      c = __builtin_amdgcn_wmma_f32_16x16x32_f16(false, a, false, bb, (short)0, c, false, false);
    }
    // fold the softmax row normalization in here (diag scaling commutes)
    #pragma unroll
    for (int r = 0; r < 8; ++r) {
      const int row = r + half * 8;
      atomicAdd(&aggS[row][n0 + l16], c[r] * invS[row]);
    }
  }
  __syncthreads();

  // ---- stage 3: LayerNorm(32) + leaky, store into concat buffer ----
  if (tid < 16) {
    const int rl = tid;
    const int i  = iblock * 16 + rl;
    float mu = 0.f;
    #pragma unroll
    for (int c = 0; c < FSQ; ++c) mu += aggS[rl][c];
    mu *= (1.f / FSQ);
    float var = 0.f;
    #pragma unroll
    for (int c = 0; c < FSQ; ++c) { const float d = aggS[rl][c] - mu; var += d * d; }
    var *= (1.f / FSQ);
    const float rstd = rsqrtf(var + LN_EPS);

    float* yrow = y + ((size_t)(b * N_ + i)) * KD + (size_t)k * FSQ;
    const float* lw = ln_w + (size_t)k * FSQ;
    const float* lb = ln_b + (size_t)k * FSQ;
    #pragma unroll
    for (int c = 0; c < FSQ; ++c)
      yrow[c] = lrelu((aggS[rl][c] - mu) * rstd * lw[c] + lb[c]);
  }
}

// ---------------------------------------------------------------------------
// Kernel D: out = y @ sq_w + sq_b + V   (f32 WMMA 16x16x4, K=160)
// grid = 1024 blocks x 256 threads; each wave -> one 16x16 output tile
// ---------------------------------------------------------------------------
__global__ __launch_bounds__(256) void squeeze_kernel(
    const float* __restrict__ y, const float* __restrict__ sq_wT,
    const float* __restrict__ sq_b, const float* __restrict__ V,
    float* __restrict__ out) {
  const int wave = threadIdx.x >> 5;
  const int lane = threadIdx.x & 31;
  const int half = lane >> 4;
  const int l16  = lane & 15;

  const int gw    = blockIdx.x * 8 + wave;  // 0..8191
  const int mtile = gw >> 4;                // 512 M-tiles
  const int ntile = gw & 15;                // 16 N-tiles
  const int m0 = mtile * 16, n0 = ntile * 16;
  const int col = n0 + l16;

  const float* arow = y + (size_t)(m0 + l16) * KD + 2 * half;
  const float* brow = sq_wT + (size_t)col * KD + 2 * half;  // transposed: pairs contiguous
  v8f c = {};
  for (int kb = 0; kb < KD; kb += 4) {
    const v2f a  = *(const v2f*)(arow + kb);
    const v2f bv = *(const v2f*)(brow + kb);
    c = __builtin_amdgcn_wmma_f32_16x16x4_f32(false, a, false, bv, (short)0, c, false, false);
  }

  const float bias = sq_b[col];
  #pragma unroll
  for (int r = 0; r < 8; ++r) {
    const int row = m0 + r + half * 8;
    out[(size_t)row * FOUT + col] = c[r] + bias + V[(size_t)row * FOUT + col];
  }
}

// ---------------------------------------------------------------------------
extern "C" void kernel_launch(void* const* d_in, const int* in_sizes, int n_in,
                              void* d_out, int out_size, void* d_ws, size_t ws_size,
                              hipStream_t stream) {
  (void)in_sizes; (void)n_in; (void)out_size; (void)ws_size;

  const float* V    = (const float*)d_in[0];
  const int*   adj  = (const int*)  d_in[1];
  const float* W    = (const float*)d_in[2];
  const float* a    = (const float*)d_in[3];
  const float* ln_w = (const float*)d_in[4];
  const float* ln_b = (const float*)d_in[5];
  const float* sq_w = (const float*)d_in[6];
  const float* sq_b = (const float*)d_in[7];
  float* out = (float*)d_out;

  // workspace layout:
  //   h    f32 [H][8192][32]        (5,242,880 B)
  //   f1   f32 [H][8192]
  //   f2   f32 [H][8192]
  //   y    f32 [8192][160]
  //   WT   f32 [H][32][256]
  //   sqT  f32 [256][160]
  //   h16t f16 [H][B][32][1024]     (2,621,440 B)   -- 32B-aligned offset
  float* h    = (float*)d_ws;
  float* f1   = h   + (size_t)H_ * ROWS * FSQ;
  float* f2   = f1  + (size_t)H_ * ROWS;
  float* y    = f2  + (size_t)H_ * ROWS;
  float* WT   = y   + (size_t)ROWS * KD;
  float* sqT  = WT  + (size_t)H_ * FSQ * FIN;
  _Float16* h16t = (_Float16*)(sqT + (size_t)FOUT * KD);

  prep_kernel<<<dim3((H_ * FSQ * FIN + FOUT * KD + 255) / 256), dim3(256), 0, stream>>>(
      W, sq_w, WT, sqT);
  head_gemm_kernel<<<dim3(H_ * 64), dim3(256), 0, stream>>>(V, WT, h, h16t);
  score_kernel<<<dim3((H_ * ROWS + 255) / 256), dim3(256), 0, stream>>>(h, a, f1, f2);
  attn_kernel<<<dim3(H_ * B_ * (N_ / 16)), dim3(256), 0, stream>>>(
      h16t, adj, f1, f2, ln_w, ln_b, y);
  squeeze_kernel<<<dim3(ROWS / 8), dim3(256), 0, stream>>>(y, sqT, sq_b, V, out);
}
